// NExpR_14903536517949
// MI455X (gfx1250) — matrix operand order repaired and checked by reading, not compiled, not run on hardware
//
#include <hip/hip_runtime.h>

typedef __attribute__((ext_vector_type(16))) _Float16 v16h;
typedef __attribute__((ext_vector_type(8)))  _Float16 v8h;
typedef __attribute__((ext_vector_type(8)))  float    v8f;

#define NB    16
#define NPTS  30000
#define HH    64
#define WW    64
#define CC    162      // 2 * L * L
#define LL    9
#define KPAD  192      // CC padded up to a multiple of 32 (6 WMMA K-chunks)
#define WAVES 4        // 128 threads / wave32
#define PTS_PER_BLOCK (WAVES * 16)
#define PI_F  3.14159265358979323846f

__global__ __launch_bounds__(128) void fourier_sample_wmma(
    const float* __restrict__ fmap, const float* __restrict__ coord,
    const float* __restrict__ basis_x, const float* __restrict__ basis_y,
    float* __restrict__ out)
{
  // Per-wave staging: Q (A-matrix, 16 points x KPAD) and P (B columns)
  __shared__ __align__(16) _Float16 Qs[WAVES][16][KPAD];
  __shared__ __align__(16) _Float16 Ps[WAVES][16][KPAD];

  const int tid  = threadIdx.x;
  const int w    = tid >> 5;     // wave in block
  const int lane = tid & 31;
  const int r    = lane & 15;    // point within tile (shared by lane, lane+16)
  const int hi16 = lane >> 4;    // upper half-wave flag

  const int tileBase = blockIdx.x * PTS_PER_BLOCK + w * 16;
  const int pid = tileBase + r;          // flat point id in [0, 480000)
  const int b   = pid / NPTS;            // batch index

  // ---- frequencies: pi * [cumsum(b[:4])-2, 0, cumsum(b[4:])] ----
  float xw[LL], yw[LL];
  {
    float cb = 0.f;
#pragma unroll
    for (int t = 0; t < 4; ++t) { cb += basis_x[t];     xw[t]     = (cb - 2.0f) * PI_F; }
    xw[4] = 0.f; cb = 0.f;
#pragma unroll
    for (int t = 0; t < 4; ++t) { cb += basis_x[4 + t]; xw[5 + t] = cb * PI_F; }
    cb = 0.f;
#pragma unroll
    for (int t = 0; t < 4; ++t) { cb += basis_y[t];     yw[t]     = (cb - 2.0f) * PI_F; }
    yw[4] = 0.f; cb = 0.f;
#pragma unroll
    for (int t = 0; t < 4; ++t) { cb += basis_y[4 + t]; yw[5 + t] = cb * PI_F; }
  }

  // ---- per-point cell + local coords ----
  const float gx = coord[2 * pid + 0];
  const float gy = coord[2 * pid + 1];
  const int hidx = (int)floorf(gx * 0.5f);
  const int widx = (int)floorf(gy * 0.5f);
  const float lx = gx - (float)hidx * 2.0f;
  const float ly = gy - (float)widx * 2.0f;

  // ---- sincos tables (angle addition avoids 162 transcendentals) ----
  float sxv[LL], cxv[LL], syv[LL], cyv[LL];
#pragma unroll
  for (int i = 0; i < LL; ++i) {
    __sincosf(lx * xw[i], &sxv[i], &cxv[i]);
    __sincosf(ly * yw[i], &syv[i], &cyv[i]);
  }

  // ---- cooperative gather of params row (648B), f32 -> f16, into LDS ----
  const long long basef = (((long long)b * HH + hidx) * WW + widx) * (long long)CC;
  if (!hi16) {
    const float2* g2 = reinterpret_cast<const float2*>(fmap + basef);   // 8B aligned
#pragma unroll 8
    for (int t = 0; t < 40; ++t) {
      float2 v = g2[t];
      Ps[w][r][2 * t]     = (_Float16)v.x;
      Ps[w][r][2 * t + 1] = (_Float16)v.y;
    }
    Ps[w][r][80] = (_Float16)fmap[basef + 80];
  } else {
    Ps[w][r][81] = (_Float16)fmap[basef + 81];
    const float2* g2 = reinterpret_cast<const float2*>(fmap + basef + 82);
#pragma unroll 8
    for (int t = 0; t < 40; ++t) {
      float2 v = g2[t];
      Ps[w][r][82 + 2 * t] = (_Float16)v.x;
      Ps[w][r][83 + 2 * t] = (_Float16)v.y;
    }
  }

  // ---- feature vector Q: sin/cos(xb_i + yb_j); i-range split across half-waves ----
  const int i0 = hi16 ? 5 : 0;
  const int i1 = hi16 ? 9 : 5;
  for (int i = i0; i < i1; ++i) {
#pragma unroll
    for (int j = 0; j < LL; ++j) {
      float s = sxv[i] * cyv[j] + cxv[i] * syv[j];   // sin(x+y)
      float c = cxv[i] * cyv[j] - sxv[i] * syv[j];   // cos(x+y)
      int idx = i * LL + j;
      Qs[w][r][idx]      = (_Float16)s;
      Qs[w][r][81 + idx] = (_Float16)c;
    }
  }
  // zero-pad K = 162..191 in both tiles
  {
    const int k0 = hi16 ? 177 : 162;
    const int k1 = hi16 ? 192 : 177;
    for (int k = k0; k < k1; ++k) {
      Qs[w][r][k] = (_Float16)0.f;
      Ps[w][r][k] = (_Float16)0.f;
    }
  }

  __syncthreads();

  // ---- 6 chained v_wmma_f32_16x16x32_f16; diag(Q x P^T) = per-point dot ----
  v8f acc = {0.f, 0.f, 0.f, 0.f, 0.f, 0.f, 0.f, 0.f};
#pragma unroll
  for (int kc = 0; kc < 6; ++kc) {
    const int k0   = kc * 32;
    const int aoff = hi16 * 8;    // A: lane<16 -> K 0-7,16-23 ; lane>=16 -> 8-15,24-31
    const int boff = hi16 * 8;    // B: lane<16 -> K 0-7,16-23 ; lane>=16 -> 8-15,24-31
    v8h a_lo = *reinterpret_cast<const v8h*>(&Qs[w][r][k0 + aoff]);
    v8h a_hi = *reinterpret_cast<const v8h*>(&Qs[w][r][k0 + 16 + aoff]);
    v8h b_lo = *reinterpret_cast<const v8h*>(&Ps[w][r][k0 + boff]);
    v8h b_hi = *reinterpret_cast<const v8h*>(&Ps[w][r][k0 + 16 + boff]);
    v16h A  = __builtin_shufflevector(a_lo, a_hi, 0,1,2,3,4,5,6,7,8,9,10,11,12,13,14,15);
    v16h Bf = __builtin_shufflevector(b_lo, b_hi, 0,1,2,3,4,5,6,7,8,9,10,11,12,13,14,15);
    acc = __builtin_amdgcn_wmma_f32_16x16x32_f16(false, A, false, Bf,
                                                 (short)0, acc, false, false);
  }

  // ---- diagonal extraction per documented C/D layout ----
  int ci = -1;
  if (lane < 8)        ci = lane;        // point p = lane      (p < 8)
  else if (lane >= 24) ci = lane - 24;   // point p = lane - 16 (p >= 8)
  if (ci >= 0) {
    float dv = acc[0];
    dv = (ci == 1) ? acc[1] : dv;
    dv = (ci == 2) ? acc[2] : dv;
    dv = (ci == 3) ? acc[3] : dv;
    dv = (ci == 4) ? acc[4] : dv;
    dv = (ci == 5) ? acc[5] : dv;
    dv = (ci == 6) ? acc[6] : dv;
    dv = (ci == 7) ? acc[7] : dv;
    out[pid] = dv;
  }
}

extern "C" void kernel_launch(void* const* d_in, const int* in_sizes, int n_in,
                              void* d_out, int out_size, void* d_ws, size_t ws_size,
                              hipStream_t stream) {
  (void)in_sizes; (void)n_in; (void)out_size; (void)d_ws; (void)ws_size;
  const float* fmap    = (const float*)d_in[0];
  const float* coord   = (const float*)d_in[1];
  const float* basis_x = (const float*)d_in[2];
  const float* basis_y = (const float*)d_in[3];
  float* out = (float*)d_out;

  const int total_pts = NB * NPTS;                  // 480000
  const int blocks = total_pts / PTS_PER_BLOCK;     // 7500 exactly
  fourier_sample_wmma<<<blocks, 128, 0, stream>>>(fmap, coord, basis_x, basis_y, out);
}
